// RCNN_37967510896701
// MI455X (gfx1250) — compile-verified, matrix operand order checked by reading, not run on hardware
//
#include <hip/hip_runtime.h>
#include <stdint.h>

#define N_BOX   10000
#define N_PAD   10016          // multiple of 32
#define W_WORDS 313            // ceil(N_PAD / 32)
#define IOU_THR 0.2f

typedef __attribute__((ext_vector_type(16))) _Float16 v16h;
typedef __attribute__((ext_vector_type(8)))  float    v8f;

// ---------------------------------------------------------------------------
// 1) Box refinement (elementwise). Writes refined boxes to d_out[0:4N) and ws.
// ---------------------------------------------------------------------------
__global__ __launch_bounds__(256) void refine_kernel(const float* __restrict__ boxes,
                                                     const float* __restrict__ deltas,
                                                     float* __restrict__ refined_ws,
                                                     float* __restrict__ out_refined) {
  int i = blockIdx.x * 256 + threadIdx.x;
  if (i >= N_BOX) return;
  float x1 = boxes[i*4+0], y1 = boxes[i*4+1], x2 = boxes[i*4+2], y2 = boxes[i*4+3];
  float cx = (x1 + x2) * 0.5f, cy = (y1 + y2) * 0.5f;
  float w  = x2 - x1,          h  = y2 - y1;
  float nx = cx + w * deltas[i*4+0];
  float ny = cy + h * deltas[i*4+1];
  float nw = w * expf(deltas[i*4+2]);
  float nh = h * expf(deltas[i*4+3]);
  float4 r = make_float4(nx - nw*0.5f, ny - nh*0.5f, nx + nw*0.5f, ny + nh*0.5f);
  ((float4*)refined_ws)[i]  = r;
  ((float4*)out_refined)[i] = r;
}

// ---------------------------------------------------------------------------
// 2) Stable descending rank via WMMA reduction-with-ones.
//    rank_i = #{ j : s_j > s_i  or (s_j == s_i and j < i) }   (exact in f32)
//    A = 16x32 f16 0/1 comparison tile, B = ones(32x16), C accumulates rank.
// ---------------------------------------------------------------------------
__global__ __launch_bounds__(256) void rank_wmma_kernel(const float* __restrict__ scores,
                                                        int* __restrict__ order) {
  __shared__ float s_lds[N_PAD];
  const int tid = threadIdx.x;
  for (int t = tid; t < N_PAD; t += 256)
    s_lds[t] = (t < N_BOX) ? scores[t] : -3.0e38f;
  __syncthreads();

  const int wave    = tid >> 5;
  const int lane    = tid & 31;
  const int rowBase = blockIdx.x * 128 + wave * 16;   // 16 rows per wave
  const int m       = lane & 15;
  const int i       = rowBase + m;                    // this lane's row (A-matrix M)
  float si = s_lds[(i < N_PAD) ? i : 0];
  if (i >= N_BOX) si = -3.0e38f;                      // pad rows: result discarded

  v16h ones;
  #pragma unroll
  for (int e = 0; e < 16; ++e) ones[e] = (_Float16)1.0f;

  v8f acc = {0.f, 0.f, 0.f, 0.f, 0.f, 0.f, 0.f, 0.f};
  // 16-bit A 16x32 layout: lanes 0-15 hold K in {0..7,16..23}, lanes 16-31 {8..15,24..31}
  const int base0 = (lane < 16) ? 0 : 8;

  for (int c = 0; c < N_PAD / 32; ++c) {
    const int j0 = c * 32;
    v16h a;
    #pragma unroll
    for (int e = 0; e < 16; ++e) {
      const int k = (e < 8) ? (base0 + e) : (base0 + 16 + (e - 8));
      const int j = j0 + k;
      const float sj = s_lds[j];
      const bool cmp = (sj > si) || ((sj == si) && (j < i));
      a[e] = cmp ? (_Float16)1.0f : (_Float16)0.0f;
    }
    acc = __builtin_amdgcn_wmma_f32_16x16x32_f16(false, a, false, ones,
                                                 (short)0, acc, false, false);
  }

  // C/D layout: lane 0 holds (N=0, M=0..7) in acc[0..7]; lane 16 holds (N=0, M=8..15)
  if (lane == 0 || lane == 16) {
    const int mb = rowBase + ((lane == 16) ? 8 : 0);
    #pragma unroll
    for (int r = 0; r < 8; ++r) {
      const int row = mb + r;
      if (row < N_BOX) {
        const int rank = (int)(acc[r] + 0.5f);
        order[rank] = row;                // ranks form a permutation -> no conflicts
      }
    }
  }
}

// ---------------------------------------------------------------------------
// 3) Gather refined boxes into sorted (descending score) order.
// ---------------------------------------------------------------------------
__global__ __launch_bounds__(256) void gather_kernel(const float* __restrict__ refined,
                                                     const int* __restrict__ order,
                                                     float* __restrict__ sorted) {
  int k = blockIdx.x * 256 + threadIdx.x;
  if (k >= N_BOX) return;
  ((float4*)sorted)[k] = ((const float4*)refined)[order[k]];
}

// ---------------------------------------------------------------------------
// 4) Suppression bitmask: mask[i][w] bit b set iff IoU(i, j=32w+b) >= thr and j > i.
//    Column boxes staged into LDS via CDNA5 async global->LDS b128 copies.
//    Block = 8 rows x 32 words (256 threads); each wave owns one row (uniform rb).
// ---------------------------------------------------------------------------
__global__ __launch_bounds__(256) void iou_mask_kernel(const float* __restrict__ sorted,
                                                       unsigned* __restrict__ mask) {
  __shared__ float4 colb[1024];                        // 16 KB of column boxes
  const int tid  = threadIdx.x;
  const int col0 = blockIdx.y * 1024;

  for (int t = tid; t < 1024; t += 256) {
    const int j = col0 + t;
    if (j < N_BOX) {
      const unsigned lds = (unsigned)(uintptr_t)(&colb[t]);
      const unsigned long long ga =
          (unsigned long long)(uintptr_t)(sorted + (size_t)j * 4);
      asm volatile("global_load_async_to_lds_b128 %0, %1, off"
                   :: "v"(lds), "v"(ga) : "memory");
    } else {
      colb[t] = make_float4(0.f, 0.f, 0.f, 0.f);       // zero-area pad -> IoU 0
    }
  }
  asm volatile("s_wait_asynccnt 0" ::: "memory");
  __syncthreads();

  const int wl = tid & 31;                             // word slot within block
  const int rl = tid >> 5;                             // row slot (== wave id)
  const int i  = blockIdx.x * 8 + rl;
  const int w  = blockIdx.y * 32 + wl;
  if (i >= N_BOX || w >= W_WORDS) return;

  const float4 rb = ((const float4*)sorted)[i];
  const float areaA = (rb.z - rb.x) * (rb.w - rb.y);
  unsigned bits = 0u;
  #pragma unroll 8
  for (int b = 0; b < 32; ++b) {
    const int bb = (b + wl) & 31;                      // rotate: spread LDS banks
    const float4 cb = colb[wl * 32 + bb];
    const int j = col0 + wl * 32 + bb;
    const float xl = fmaxf(rb.x, cb.x), yt = fmaxf(rb.y, cb.y);
    const float xr = fminf(rb.z, cb.z), yb = fminf(rb.w, cb.w);
    const float inter = fmaxf(xr - xl, 0.f) * fmaxf(yb - yt, 0.f);
    const float areaB = (cb.z - cb.x) * (cb.w - cb.y);
    const float uni = areaA + areaB - inter;
    const bool sup = (inter >= IOU_THR * uni) && (j > i);   // union > 0 always
    bits |= sup ? (1u << bb) : 0u;
  }
  mask[(size_t)i * W_WORDS + w] = bits;                // wave-coalesced store
}

// ---------------------------------------------------------------------------
// 5) Sequential greedy scan (one block). removed bitmap lives in LDS; suppressed
//    rows cost no memory traffic and no barriers. Tail scatters keep-mask to
//    original ordering as 0.0/1.0 floats.
// ---------------------------------------------------------------------------
__global__ __launch_bounds__(512) void nms_scan_kernel(const unsigned* __restrict__ mask,
                                                       const int* __restrict__ order,
                                                       float* __restrict__ out_keep) {
  __shared__ unsigned removed[W_WORDS];
  __shared__ unsigned keepbits[W_WORDS];
  const int tid = threadIdx.x;
  for (int t = tid; t < W_WORDS; t += 512) { removed[t] = 0u; keepbits[t] = 0u; }
  __syncthreads();

  for (int w = 0; w < W_WORDS; ++w) {
    unsigned rm = removed[w];                          // uniform across threads
    const int bmax = (w * 32 + 32 <= N_BOX) ? 32 : (N_BOX - w * 32);
    for (int b = 0; b < bmax; ++b) {
      if (rm & (1u << b)) continue;                    // suppressed: free skip
      const int i = w * 32 + b;
      if (tid == 0) keepbits[w] |= (1u << b);
      const unsigned* row = mask + (size_t)i * W_WORDS;
      for (int t = w + tid; t < W_WORDS; t += 512)     // words < w are all-zero
        removed[t] |= row[t];
      __syncthreads();
      rm = removed[w];                                 // may have gained bits > b
    }
  }
  __syncthreads();

  for (int k = tid; k < N_BOX; k += 512) {
    const int orig = order[k];
    out_keep[orig] = ((keepbits[k >> 5] >> (k & 31)) & 1u) ? 1.0f : 0.0f;
  }
}

// ---------------------------------------------------------------------------
extern "C" void kernel_launch(void* const* d_in, const int* in_sizes, int n_in,
                              void* d_out, int out_size, void* d_ws, size_t ws_size,
                              hipStream_t stream) {
  const float* boxes  = (const float*)d_in[0];
  const float* deltas = (const float*)d_in[1];
  const float* scores = (const float*)d_in[2];
  float* out = (float*)d_out;

  char* ws = (char*)d_ws;
  float*    refined = (float*)(ws + 0);           // 160,000 B
  float*    sorted  = (float*)(ws + 163840);      // 160,000 B
  int*      order   = (int*)  (ws + 327680);      //  40,000 B
  unsigned* mask    = (unsigned*)(ws + 368640);   // 12,520,000 B

  refine_kernel<<<(N_BOX + 255) / 256, 256, 0, stream>>>(boxes, deltas, refined, out);
  rank_wmma_kernel<<<(N_BOX + 127) / 128, 256, 0, stream>>>(scores, order);
  gather_kernel<<<(N_BOX + 255) / 256, 256, 0, stream>>>(refined, order, sorted);
  iou_mask_kernel<<<dim3((N_BOX + 7) / 8, (N_PAD + 1023) / 1024), 256, 0, stream>>>(sorted, mask);
  nms_scan_kernel<<<1, 512, 0, stream>>>(mask, order, out + (size_t)N_BOX * 4);
}
